// RNN_16767552324102
// MI455X (gfx1250) — matrix operand order repaired and checked by reading, compiled, not run on hardware
//
#include <hip/hip_runtime.h>

// ---------------------------------------------------------------------------
// 2-layer ReLU RNN (B=64, T=256, D=512, H=1024, C=1000) for gfx1250 (MI455X).
//   * All matmuls via v_wmma_f32_16x16x32_f16 (f16 in, f32 accumulate).
//   * Input projections for ALL timesteps precomputed as register-blocked
//     (2Mx4N tiles/wave) WMMA GEMMs.
//   * Sequential recurrence h = relu(xproj_t + h @ Whh^T): persistent 16-block
//     kernel; each block stages its 128KB W_hh slice into LDS once (320KB/WGP
//     on CDNA5), B operands then come from conflict-free ds_load_b128 every
//     step; the h A-fragment global load is software-pipelined; grid-wide
//     release/acquire barrier between timesteps.
//   * Weights kept [N][K] row-major == B-operand native layout (no transpose).
// ---------------------------------------------------------------------------

typedef __attribute__((ext_vector_type(16))) _Float16 v16h;
typedef __attribute__((ext_vector_type(8)))  _Float16 v8h;
typedef __attribute__((ext_vector_type(8)))  float    v8f;

union FragU { v16h v; v8h h[2]; };

#define LDS_STRIDE 1032   // 1024 + 8 halves pad -> 516 dwords = 4 (mod 64 banks)

// A-matrix 16x32 f16 layout: lane 0-15 row=lane, halves0..15=K 0..15 ;
// lanes 16-31: K 16..31 (contiguous, matches B).
__device__ __forceinline__ v16h load_frag_a(const _Float16* base, int stride,
                                            int row0, int k0) {
  const int lane = threadIdx.x & 31;
  const int r  = row0 + (lane & 15);
  const int kb = (lane >> 4) << 4;              // 0 or 16
  const _Float16* p = base + (size_t)r * stride + k0 + kb;
  FragU f;
  f.h[0] = *(const v8h*)(p);                    // K = kb .. kb+7
  f.h[1] = *(const v8h*)(p + 8);                // K = kb+8 .. kb+15
  return f.v;
}

// B-matrix 32x16 f16 layout: lane holds one column (row of W), 16 contiguous K:
// lanes 0-15 -> K 0..15, lanes 16-31 -> K 16..31.
__device__ __forceinline__ v16h load_frag_b(const _Float16* base, int stride,
                                            int row0, int k0) {
  const int lane = threadIdx.x & 31;
  const int r  = row0 + (lane & 15);
  const int kb = (lane >> 4) << 4;              // 0 or 16
  const _Float16* p = base + (size_t)r * stride + k0 + kb;
  FragU f;
  f.h[0] = *(const v8h*)(p);
  f.h[1] = *(const v8h*)(p + 8);
  return f.v;
}

// Same B fragment, sourced from the padded LDS copy of the W slice.
__device__ __forceinline__ v16h load_frag_b_lds(const _Float16* base,
                                                int row0, int k0) {
  const int lane = threadIdx.x & 31;
  const int r  = row0 + (lane & 15);
  const int kb = (lane >> 4) << 4;
  const _Float16* p = base + r * LDS_STRIDE + k0 + kb;
  FragU f;
  f.h[0] = *(const v8h*)(p);
  f.h[1] = *(const v8h*)(p + 8);
  return f.v;
}

__device__ __forceinline__ v8f wmma_f16(v16h a, v16h b, v8f c) {
  // (neg_a, A, neg_b, B, c_mod, C, reuse_a, reuse_b)
  return __builtin_amdgcn_wmma_f32_16x16x32_f16(false, a, false, b,
                                                (short)0, c, false, false);
}

// --------------------------- conversion kernels ----------------------------

__global__ void cvt_f32_to_f16(const float* __restrict__ in,
                               _Float16* __restrict__ out, int n) {
  int i = blockIdx.x * 256 + threadIdx.x;
  if (i < n) out[i] = (_Float16)in[i];
}

// w_fc [1000,1024] -> padded f16 [1024,1024] (rows 1000..1023 zero)
__global__ void cvt_wfc(const float* __restrict__ in,
                        _Float16* __restrict__ out) {
  int i = blockIdx.x * 256 + threadIdx.x;      // over 1024*1024
  int r = i >> 10;
  int c = i & 1023;
  out[i] = (r < 1000) ? (_Float16)in[(size_t)r * 1024 + c] : (_Float16)0.f;
}

__global__ void init_counters(int* bar) {
  if (threadIdx.x < 2) bar[threadIdx.x] = 0;
}

// ------------------------- input-projection GEMM ---------------------------
// out[t][b][n] = sum_k A[b*256+t][k] * W[n][k]
// A: [16384, K] f16 row-major (rows = b*T+t), W: [1024, K] f16 row-major.
// Register blocking: each wave computes 2 M-tiles x 4 N-tiles (32x64 output).
// Block = 8 waves stacked in M (256 rows x 64 cols). grid(16, 64), block 256.
__global__ void gemm_xproj(const _Float16* __restrict__ A,
                           const _Float16* __restrict__ W,
                           float* __restrict__ out, int K) {
  const int wave = threadIdx.x >> 5;
  const int n0 = blockIdx.x * 64;                    // 4 n-tiles
  const int m0 = (blockIdx.y * 8 + wave) * 32;       // 2 m-tiles
  v8f acc[2][4] = {};
#pragma unroll 2
  for (int k0 = 0; k0 < K; k0 += 32) {
    v16h a0 = load_frag_a(A, K, m0,      k0);
    v16h a1 = load_frag_a(A, K, m0 + 16, k0);
#pragma unroll
    for (int j = 0; j < 4; ++j) {
      v16h b = load_frag_b(W, K, n0 + 16 * j, k0);
      acc[0][j] = wmma_f16(a0, b, acc[0][j]);
      acc[1][j] = wmma_f16(a1, b, acc[1][j]);
    }
  }
  const int lane = threadIdx.x & 31;
#pragma unroll
  for (int i2 = 0; i2 < 2; ++i2) {
#pragma unroll
    for (int j = 0; j < 4; ++j) {
      int nn = n0 + 16 * j + (lane & 15);
      int mbase = m0 + 16 * i2 + ((lane >> 4) << 3);
#pragma unroll
      for (int i = 0; i < 8; ++i) {
        int r = mbase + i;               // r = b*256 + t
        int t = r & 255, b = r >> 8;
        out[((size_t)t * 64 + b) * 1024 + nn] = acc[i2][j][i];
      }
    }
  }
}

// ---------------------------- recurrent kernel -----------------------------
// h_t = relu(xproj[t] + h_{t-1} @ Whh^T), 256 steps with grid-wide barrier.
// grid = 16 blocks x 256 threads (8 waves) -> 128 waves, each 1 M-tile x
// 2 N-tiles. Block b owns output columns [64b, 64b+64): its W slice (64 rows
// x 1024 f16 = 128KB) is staged into LDS once and reused for all 256 steps.
// LAYER1==0: hist = full history  [b*256+t][1024] f16 (feeds next GEMM).
// LAYER1==1: hist = ping-pong [2][64][1024] f16 (only last state needed).
template <int LAYER1>
__global__ void rnn_recurrent(const float* __restrict__ xproj,
                              const _Float16* __restrict__ W,
                              _Float16* __restrict__ hist,
                              int* __restrict__ bar) {
  __shared__ _Float16 Wlds[64 * LDS_STRIDE];   // 132,096 B (<320KB/WGP)

  const int wave = threadIdx.x >> 5;
  const int gw = blockIdx.x * 8 + wave;        // 0..127
  const int m0 = (gw & 3) * 16;                // batch tile
  const int n0 = (gw >> 2) * 32;               // 2 hidden-unit tiles (global)
  const int nl0 = n0 - blockIdx.x * 64;        // local row in LDS slice (0|32)
  const int lane = threadIdx.x & 31;
  const int nn0 = n0 + (lane & 15);
  const int mb = m0 + ((lane >> 4) << 3);

  // one-time stage of this block's W slice into LDS (8-half vector copies)
  {
    const _Float16* Wblk = W + (size_t)blockIdx.x * 64 * 1024;
    for (int e = threadIdx.x; e < 64 * 128; e += 256) {
      int row = e >> 7;                 // 128 x 8-half chunks per row
      int col = (e & 127) << 3;
      *(v8h*)&Wlds[row * LDS_STRIDE + col] =
          *(const v8h*)&Wblk[(size_t)row * 1024 + col];
    }
  }
  __syncthreads();

  for (int t = 0; t < 256; ++t) {
    // prefetch the additive input-projection terms (overlap with WMMA stream)
    float xadd[2][8];
#pragma unroll
    for (int i = 0; i < 8; ++i) {
      const float* xp = xproj + ((size_t)t * 64 + mb + i) * 1024;
      xadd[0][i] = xp[nn0];
      xadd[1][i] = xp[nn0 + 16];
    }
    v8f acc0 = {}, acc1 = {};
    if (t > 0) {
      const _Float16* hbase;
      int hstride;
      if (!LAYER1) { hbase = hist + (size_t)(t - 1) * 1024; hstride = 256 * 1024; }
      else         { hbase = hist + (size_t)((t - 1) & 1) * 64 * 1024; hstride = 1024; }
      // software pipeline: global A fragment for k+32 in flight during WMMAs
      v16h a = load_frag_a(hbase, hstride, m0, 0);
#pragma unroll 2
      for (int k0 = 32; k0 < 1024; k0 += 32) {
        v16h an = load_frag_a(hbase, hstride, m0, k0);
        v16h b0 = load_frag_b_lds(Wlds, nl0,      k0 - 32);
        v16h b1 = load_frag_b_lds(Wlds, nl0 + 16, k0 - 32);
        acc0 = wmma_f16(a, b0, acc0);
        acc1 = wmma_f16(a, b1, acc1);
        a = an;
      }
      v16h b0 = load_frag_b_lds(Wlds, nl0,      1024 - 32);
      v16h b1 = load_frag_b_lds(Wlds, nl0 + 16, 1024 - 32);
      acc0 = wmma_f16(a, b0, acc0);
      acc1 = wmma_f16(a, b1, acc1);
    }
    // epilogue: add input projection, ReLU, store f16 state
    _Float16* hout;
    size_t ostride;
    if (!LAYER1) { hout = hist + (size_t)t * 1024; ostride = (size_t)256 * 1024; }
    else         { hout = hist + (size_t)(t & 1) * 64 * 1024; ostride = 1024; }
#pragma unroll
    for (int i = 0; i < 8; ++i) {
      float v0 = acc0[i] + xadd[0][i];
      float v1 = acc1[i] + xadd[1][i];
      v0 = v0 > 0.f ? v0 : 0.f;
      v1 = v1 > 0.f ? v1 : 0.f;
      _Float16* hp = hout + (size_t)(mb + i) * ostride;
      hp[nn0]      = (_Float16)v0;
      hp[nn0 + 16] = (_Float16)v1;
    }
    // grid-wide barrier between timesteps (all 16 blocks co-resident)
    if (t != 255) {
      __threadfence();                          // release our h stores
      __syncthreads();
      if (threadIdx.x == 0) {
        __hip_atomic_fetch_add(bar, 1, __ATOMIC_RELEASE, __HIP_MEMORY_SCOPE_AGENT);
        const int target = (t + 1) * (int)gridDim.x;
        while (__hip_atomic_load(bar, __ATOMIC_ACQUIRE,
                                 __HIP_MEMORY_SCOPE_AGENT) < target) {
          __builtin_amdgcn_s_sleep(1);
        }
      }
      __syncthreads();
      __threadfence();                          // acquire others' h stores
    }
  }
}

// -------------------------------- FC kernel --------------------------------
// out[b][c] = sum_k h[b][k] * Wfc[c][k] ; Wfc padded to 1024 rows.
// grid 64 (N-tiles), block 128 (4 waves = 4 M-tiles).
__global__ void fc_kernel(const _Float16* __restrict__ h,
                          const _Float16* __restrict__ Wfc,
                          float* __restrict__ out) {
  const int n0 = blockIdx.x * 16;
  const int m0 = (threadIdx.x >> 5) * 16;
  v8f c = {};
#pragma unroll 4
  for (int k0 = 0; k0 < 1024; k0 += 32) {
    v16h a = load_frag_a(h, 1024, m0, k0);
    v16h b = load_frag_b(Wfc, 1024, n0, k0);
    c = wmma_f16(a, b, c);
  }
  const int lane = threadIdx.x & 31;
  const int nn = n0 + (lane & 15);
  const int mb = m0 + ((lane >> 4) << 3);
  if (nn < 1000) {
#pragma unroll
    for (int i = 0; i < 8; ++i)
      out[(size_t)(mb + i) * 1000 + nn] = c[i];
  }
}

// ------------------------------- launcher ----------------------------------

extern "C" void kernel_launch(void* const* d_in, const int* in_sizes, int n_in,
                              void* d_out, int out_size, void* d_ws, size_t ws_size,
                              hipStream_t stream) {
  const float* x     = (const float*)d_in[0];   // [64,256,512]
  const float* w_ih0 = (const float*)d_in[1];   // [1024,512]
  const float* w_hh0 = (const float*)d_in[2];   // [1024,1024]
  const float* w_ih1 = (const float*)d_in[3];   // [1024,1024]
  const float* w_hh1 = (const float*)d_in[4];   // [1024,1024]
  const float* w_fc  = (const float*)d_in[5];   // [1000,1024]
  float* out = (float*)d_out;                   // [64,1000]

  // workspace carve-up (~127 MB)
  char* p = (char*)d_ws;
  auto carve = [&](size_t bytes) {
    void* r = (void*)p;
    p += (bytes + 255) & ~(size_t)255;
    return r;
  };
  int*      bar    = (int*)carve(256);
  _Float16* x16    = (_Float16*)carve((size_t)16384 * 512 * 2);
  _Float16* wih0h  = (_Float16*)carve((size_t)1024 * 512 * 2);
  _Float16* whh0h  = (_Float16*)carve((size_t)1024 * 1024 * 2);
  _Float16* wih1h  = (_Float16*)carve((size_t)1024 * 1024 * 2);
  _Float16* whh1h  = (_Float16*)carve((size_t)1024 * 1024 * 2);
  _Float16* wfch   = (_Float16*)carve((size_t)1024 * 1024 * 2);
  _Float16* h1     = (_Float16*)carve((size_t)16384 * 1024 * 2); // layer0 history
  _Float16* hping  = (_Float16*)carve((size_t)2 * 64 * 1024 * 2); // layer1 state
  float*    xproj  = (float*)carve((size_t)256 * 64 * 1024 * 4);  // reused L0/L1

  init_counters<<<1, 32, 0, stream>>>(bar);

  // f32 -> f16 conversions
  cvt_f32_to_f16<<<32768, 256, 0, stream>>>(x,     x16,   16384 * 512);
  cvt_f32_to_f16<<<2048,  256, 0, stream>>>(w_ih0, wih0h, 1024 * 512);
  cvt_f32_to_f16<<<4096,  256, 0, stream>>>(w_hh0, whh0h, 1024 * 1024);
  cvt_f32_to_f16<<<4096,  256, 0, stream>>>(w_ih1, wih1h, 1024 * 1024);
  cvt_f32_to_f16<<<4096,  256, 0, stream>>>(w_hh1, whh1h, 1024 * 1024);
  cvt_wfc<<<4096, 256, 0, stream>>>(w_fc, wfch);

  // layer 0: xproj0 = x @ w_ih0^T  (K=512), then recurrence -> h1 history
  gemm_xproj<<<dim3(16, 64), 256, 0, stream>>>(x16, wih0h, xproj, 512);
  rnn_recurrent<0><<<16, 256, 0, stream>>>(xproj, whh0h, h1, bar + 0);

  // layer 1: xproj1 = h1 @ w_ih1^T (K=1024), then recurrence -> ping-pong
  gemm_xproj<<<dim3(16, 64), 256, 0, stream>>>(h1, wih1h, xproj, 1024);
  rnn_recurrent<1><<<16, 256, 0, stream>>>(xproj, whh1h, hping, bar + 1);

  // final h is in ping buffer 1 (t=255 -> 255&1)
  fc_kernel<<<64, 128, 0, stream>>>(hping + (size_t)64 * 1024, wfch, out);
}
